// TransformerEmbedding_15865609191705
// MI455X (gfx1250) — compile-verified, matrix operand order checked by reading, not run on hardware
//
#include <hip/hip_runtime.h>

typedef __attribute__((ext_vector_type(16))) _Float16 v16h;
typedef __attribute__((ext_vector_type(8)))  _Float16 v8h;
typedef __attribute__((ext_vector_type(8)))  float    v8f;

#define D_MODEL 1024
#define N_TYPES 21
#define KPAD    32
#define N_SEQS  64
#define LN10000 9.210340371976184f
#define HALF_PI 1.5707963267948966f
#define INV_2PI 0.15915494309189535f

// ---------------------------------------------------------------------------
// Kernel 1: segment offsets via binary search over the sorted batch array.
// offs[s] = first index with batch[i] >= s ; offs[64] = n_tokens.
// ---------------------------------------------------------------------------
__global__ void te_offsets_kernel(const int* __restrict__ batch, int n_tokens,
                                  int* __restrict__ offs) {
    int s = threadIdx.x;
    if (s < N_SEQS) {
        int lo = 0, hi = n_tokens;
        while (lo < hi) {
            int mid = (lo + hi) >> 1;
            if (batch[mid] < s) lo = mid + 1; else hi = mid;
        }
        offs[s] = lo;
    } else if (s == N_SEQS) {
        offs[N_SEQS] = n_tokens;
    }
}

// ---------------------------------------------------------------------------
// Kernel 2: per-token scatter metadata:
// rowinfo[n] = { (seg*max_len+pos)*1024, float_bits((float)pos) }.
// ---------------------------------------------------------------------------
__global__ void te_rowinfo_kernel(const int* __restrict__ batch,
                                  const int* __restrict__ offs,
                                  int2* __restrict__ rowinfo,
                                  int n_tokens, int max_len) {
    int n = blockIdx.x * blockDim.x + threadIdx.x;
    if (n >= n_tokens) return;
    int seg = batch[n];
    int pos = n - offs[seg];
    rowinfo[n] = make_int2((seg * max_len + pos) * D_MODEL,
                           __float_as_int((float)pos));
}

// ---------------------------------------------------------------------------
// Kernel 3: pack f32 [nrows][21] -> f16 [nrows][32], zero-padded K=21..31.
// Load index clamped (always in-bounds) + v_cndmask -> branchless.
// ---------------------------------------------------------------------------
__global__ void te_pack_kernel(const float* __restrict__ src,
                               _Float16* __restrict__ dst, int nrows) {
    int tid = blockIdx.x * blockDim.x + threadIdx.x;
    if (tid >= nrows * KPAD) return;
    int n = tid >> 5;
    int k = tid & 31;
    int kc = (k < N_TYPES) ? k : (N_TYPES - 1);
    float v = src[n * N_TYPES + kc];
    dst[tid] = (k < N_TYPES) ? (_Float16)v : (_Float16)0.0f;
}

// ---------------------------------------------------------------------------
// Kernel 4: per-channel epilogue table:
// chan[d] = { bias[d]*sqrt(d_model), PE inverse frequency in revolutions }.
// ---------------------------------------------------------------------------
__global__ void te_chan_kernel(const float* __restrict__ bias,
                               float2* __restrict__ chan) {
    int d = blockIdx.x * blockDim.x + threadIdx.x;
    if (d >= D_MODEL) return;
    float invf = __expf((float)(d & ~1) * (-LN10000 / (float)D_MODEL));
    chan[d] = make_float2(bias[d] * 32.0f, invf * INV_2PI);
}

// ---------------------------------------------------------------------------
// Kernel 5: PE into *padding* rows only (valid rows fully written by the WMMA
// kernel -> disjoint stores), plus the mask. Cold path.
// ---------------------------------------------------------------------------
__global__ void te_fill_kernel(float* __restrict__ out, float* __restrict__ mask,
                               const int* __restrict__ offs, int max_len) {
    const int qPerRow = D_MODEL / 4;
    long tid = (long)blockIdx.x * blockDim.x + threadIdx.x;
    long total = (long)N_SEQS * max_len * qPerRow;
    if (tid >= total) return;
    int q = (int)(tid % qPerRow);
    int p = (int)((tid / qPerRow) % max_len);
    int s = (int)(tid / ((long)qPerRow * max_len));

    int cnt = offs[s + 1] - offs[s];
    if (q == 0)
        mask[(long)s * max_len + p] = (p < cnt) ? 1.0f : 0.0f;
    if (p < cnt) return;                      // valid row: WMMA kernel owns it

    int d0 = q * 4;
    float4 v;
#pragma unroll
    for (int e = 0; e < 4; ++e) {
        int d = d0 + e;
        float invf  = __expf((float)(d & ~1) * (-LN10000 / (float)D_MODEL));
        float phase = (d & 1) ? HALF_PI : 0.0f;
        ((float*)&v)[e] = __sinf((float)p * invf + phase);
    }
    *(float4*)(out + ((long)s * max_len + p) * D_MODEL + d0) = v;
}

// ---------------------------------------------------------------------------
// Kernel 6: fused WMMA GEMM + bias + *32 + PE + scatter. One wave: 16 tokens x
// 64 channels = 4x v_wmma_f32_16x16x32_f16 sharing one A tile. Operands from
// K-padded f16 arrays: per lane, elements 0-7 = halves K=8H..8H+7, elements
// 8-15 = K=16+8H..16+8H+7 -> two b128 loads each, zero branches. Epilogue:
// hardware-native v_sin (revolutions), unsigned 32-bit store offsets so
// stores use the saddr + zext(voffset) form.
// ---------------------------------------------------------------------------
__global__ void __launch_bounds__(256)
te_gemm_scatter_kernel(const _Float16* __restrict__ xpad,
                       const _Float16* __restrict__ wpad,
                       const float2* __restrict__ chan,
                       const int2* __restrict__ rowinfo,
                       float* __restrict__ out) {
    int lane   = threadIdx.x & 31;
    int wave   = blockIdx.x * 8 + (threadIdx.x >> 5);
    int mTile  = wave >> 4;                 // 16 d-groups of 64 channels
    int dGroup = wave & 15;
    int tBase  = mTile << 4;
    int dBase  = dGroup << 6;
    int H = lane >> 4;
    int r = lane & 15;

    const v8h* xr = (const v8h*)(xpad + (tBase + r) * KPAD);
    v8h alo = xr[H], ahi = xr[2 + H];
    v16h a = __builtin_shufflevector(alo, ahi,
                                     0,1,2,3,4,5,6,7,8,9,10,11,12,13,14,15);

    v8f   c[4];
    float bs32[4], invf[4];
#pragma unroll
    for (int t = 0; t < 4; ++t) {
        const v8h* wr = (const v8h*)(wpad + (dBase + 16 * t + r) * KPAD);
        v8h blo = wr[H], bhi = wr[2 + H];
        v16h bm = __builtin_shufflevector(blo, bhi,
                                          0,1,2,3,4,5,6,7,8,9,10,11,12,13,14,15);
        v8f z = {};
        c[t] = __builtin_amdgcn_wmma_f32_16x16x32_f16(
            /*neg_a=*/false, a, /*neg_b=*/false, bm,
            /*c_mod=*/(short)0, z, /*reuse_a=*/false, /*reuse_b=*/false);
        float2 ch = chan[dBase + 16 * t + r];
        bs32[t] = ch.x;                      // bias * 32
        invf[t] = ch.y;                      // inverse freq in revolutions
    }

    // d parity == r parity for all 4 subtiles -> phase is lane-constant.
    // cos(x) = sin(x + pi/2) -> +0.25 revolutions.
    float phase = (r & 1) ? 0.25f : 0.0f;

    // C/D layout: vgpr v holds M = v (lanes 0-15) or M = v+8 (lanes 16-31), N = r.
#pragma unroll
    for (int v = 0; v < 8; ++v) {
        int  tok = tBase + (H << 3) + v;
        int2 rp  = rowinfo[tok];             // {rowbase, float_bits(pos)}
        float fpos  = __int_as_float(rp.y);
        unsigned off = (unsigned)rp.x + (unsigned)(dBase + r);
#pragma unroll
        for (int t = 0; t < 4; ++t) {
            float pe = __builtin_amdgcn_sinf(fmaf(fpos, invf[t], phase));
            out[off + 16u * t] = fmaf(c[t][v], 32.0f, bs32[t]) + pe;
        }
    }
}

// ---------------------------------------------------------------------------
extern "C" void kernel_launch(void* const* d_in, const int* in_sizes, int n_in,
                              void* d_out, int out_size, void* d_ws, size_t ws_size,
                              hipStream_t stream) {
    (void)n_in; (void)ws_size;
    const float* x     = (const float*)d_in[0];
    const float* W     = (const float*)d_in[1];
    const float* b     = (const float*)d_in[2];
    const int*   batch = (const int*)d_in[3];
    int n_tokens = in_sizes[3];

    // out_size = 64*max_len*1024 (tensor) + 64*max_len (mask); recover max_len.
    int max_len;
    if (out_size % (N_SEQS * (D_MODEL + 1)) == 0)
        max_len = out_size / (N_SEQS * (D_MODEL + 1));
    else
        max_len = out_size / (N_SEQS * D_MODEL);

    float* out  = (float*)d_out;
    float* mask = out + (long)N_SEQS * max_len * D_MODEL;

    // Workspace layout (256B-aligned chunks):
    char* ws = (char*)d_ws;
    int*      offs    = (int*)ws;                                   // 65 ints
    int2*     rowinfo = (int2*)(ws + 256);                          // n_tokens*8 B
    _Float16* xpad    = (_Float16*)(ws + 256 + (size_t)n_tokens * 8);
    _Float16* wpad    = xpad + (size_t)n_tokens * KPAD;             // 1024*32*2 B
    float2*   chan    = (float2*)(wpad + (size_t)D_MODEL * KPAD);   // 1024*8 B

    te_offsets_kernel<<<1, 128, 0, stream>>>(batch, n_tokens, offs);

    te_rowinfo_kernel<<<(n_tokens + 255) / 256, 256, 0, stream>>>(
        batch, offs, rowinfo, n_tokens, max_len);

    te_pack_kernel<<<(n_tokens * KPAD + 255) / 256, 256, 0, stream>>>(
        x, xpad, n_tokens);
    te_pack_kernel<<<(D_MODEL * KPAD + 255) / 256, 256, 0, stream>>>(
        W, wpad, D_MODEL);
    te_chan_kernel<<<D_MODEL / 256, 256, 0, stream>>>(b, chan);

    long fill_threads = (long)N_SEQS * max_len * (D_MODEL / 4);
    int  fill_blocks  = (int)((fill_threads + 255) / 256);
    te_fill_kernel<<<fill_blocks, 256, 0, stream>>>(out, mask, offs, max_len);

    int nWaves = (n_tokens / 16) * (D_MODEL / 64);   // 2048 * 16 = 32768
    int blocks = nWaves / 8;                          // 8 waves / 256-thread block
    te_gemm_scatter_kernel<<<blocks, 256, 0, stream>>>(xpad, wpad, chan, rowinfo, out);
}
